// DenseEquivariantShiftModule_88579405512993
// MI455X (gfx1250) — compile-verified, matrix operand order checked by reading, not run on hardware
//
#include <hip/hip_runtime.h>

typedef __attribute__((ext_vector_type(16))) _Float16 v16h;
typedef __attribute__((ext_vector_type(8)))  _Float16 v8h;
typedef __attribute__((ext_vector_type(2)))  __fp16   v2h;   // matches cvt_pkrtz return type
typedef __attribute__((ext_vector_type(8)))  float    v8f;
typedef __attribute__((ext_vector_type(4)))  float    v4f;

#define BB 2
#define NN 512
#define FF 128
#define NB 4

// ---- workspace layout (byte offsets, all 16B aligned) ----
// wfrags:  68 frags * 1024 halves * 2B = 139264   (rel W1^T/W2^T/W3^T in WMMA A layout)
// biases:  272 floats (b1[128], b2[128], b3 padded to 16) = 1088  -> contiguous w/ wfrags
// invNa:   4 floats { 1/na[b0], 1/na[b1], 1/na^2[b0], 1/na^2[b1] }
// pwpart:  B*N*3 floats
#define WS_WFRAGS 0
#define WS_BIAS   139264
#define WS_INVNA  140352
#define WS_PWPART 140416

union H16 { v16h v; v8h h8[2]; v2h h2[8]; };

__device__ __forceinline__ v16h load_frag(const _Float16* wlds, int f, int lane) {
  const v8h a0 = *(const v8h*)(wlds + f*1024 + lane*8);
  const v8h a1 = *(const v8h*)(wlds + f*1024 + 512 + lane*8);
  return __builtin_shufflevector(a0, a1, 0,1,2,3,4,5,6,7,8,9,10,11,12,13,14,15);
}

// ---------------------------------------------------------------------------
// Prep: pack transposed rel weights into WMMA A-fragments.
// A tile (M=16 hidden, K=32 in):  A[m][k] = W[k_in][m_out]  (W row-major [K][Cout])
// Frag storage: 2 blocks of 1024B; block p holds lane's 8 halves:
//   lanes 0-15 : halves = K {p*16+0..7},  lanes 16-31 : K {p*16+8..15}
// ---------------------------------------------------------------------------
__global__ void prep_kernel(const float* __restrict__ W1, const float* __restrict__ W2,
                            const float* __restrict__ W3,
                            const float* __restrict__ b1, const float* __restrict__ b2,
                            const float* __restrict__ b3,
                            const unsigned char* __restrict__ maskb,
                            _Float16* __restrict__ wfrags, float* __restrict__ biases,
                            float* __restrict__ invNa) {
  const int fid = blockIdx.x;
  const int tid = threadIdx.x;
  if (fid < 68) {
    const float* W; int cout, t, s;
    if (fid < 32)      { W = W1; cout = 128; t = fid >> 2;      s = fid & 3; }
    else if (fid < 64) { W = W2; cout = 128; t = (fid-32) >> 2; s = (fid-32) & 3; }
    else               { W = W3; cout = NB;  t = 0;             s = fid - 64; }
    for (int e = tid; e < 1024; e += 256) {
      const int p = e >> 9, rem = e & 511, l = rem >> 3, j = rem & 7;
      const int m   = l & 15;
      const int kk  = p*16 + ((l >> 4) << 3) + j;   // K within 32-step
      const int k   = s*32 + kk;
      const int col = t*16 + m;
      const float v = (col < cout) ? W[k*cout + col] : 0.0f;
      wfrags[fid*1024 + e] = (_Float16)v;
    }
  } else if (fid == 68) {
    for (int e = tid; e < 272; e += 256) {
      float v;
      if (e < 128)      v = b1[e];
      else if (e < 256) v = b2[e-128];
      else              v = (e-256 < NB) ? b3[e-256] : 0.0f;
      biases[e] = v;
    }
  } else {
    if (tid < BB) {
      int cnt = 0;
      for (int j = 0; j < NN; ++j) cnt += (maskb[tid*NN + j] == 0) ? 1 : 0;
      const float na = (float)cnt;
      invNa[tid]     = 1.0f / na;
      invNa[2 + tid] = 1.0f / (na*na);
    }
  }
}

// ---------------------------------------------------------------------------
// Pointwise MLP (tiny: 1024 rows) -> pwpart[b,i,d] = (1/na) * sum_k scale_k*basis
// ---------------------------------------------------------------------------
__global__ void pw_kernel(const float* __restrict__ pwF, const float* __restrict__ pwBas,
                          const float* __restrict__ W1, const float* __restrict__ b1,
                          const float* __restrict__ W2, const float* __restrict__ b2,
                          const float* __restrict__ W3, const float* __restrict__ b3,
                          const float* __restrict__ invNa, float* __restrict__ pwpart) {
  __shared__ float xs[FF];
  __shared__ float hs[FF];
  __shared__ float scale[NB];
  const int bi = blockIdx.x;        // b*N + i
  const int b  = bi / NN;
  const int tid = threadIdx.x;      // 128 threads
  xs[tid] = pwF[(size_t)bi*FF + tid];
  __syncthreads();
  float a = b1[tid];
  for (int f = 0; f < FF; ++f) a += xs[f] * W1[f*FF + tid];
  hs[tid] = fmaxf(a, 0.0f);
  __syncthreads();
  float a2 = b2[tid];
  for (int f = 0; f < FF; ++f) a2 += hs[f] * W2[f*FF + tid];
  a2 = fmaxf(a2, 0.0f);
  __syncthreads();
  xs[tid] = a2;
  __syncthreads();
  if (tid < NB) {
    float s3 = b3[tid];
    for (int c = 0; c < FF; ++c) s3 += xs[c] * W3[c*NB + tid];
    scale[tid] = s3;
  }
  __syncthreads();
  if (tid < 3) {
    float acc = 0.0f;
    for (int k = 0; k < NB; ++k)
      acc += scale[k] * pwBas[((size_t)bi*NB + k)*3 + tid];
    pwpart[bi*3 + tid] = invNa[b] * acc;
  }
}

// ---------------------------------------------------------------------------
// Main: one workgroup per (b,i). 8 waves x 4 tiles of 16 rows j.
// Computes H^T = W^T X^T so D tiles feed the next layer's B frags in-register.
// Per K-step: issue all 16 ds_load_b128 (A frags) first, then 8 WMMAs -> the
// DS pipe stays full and waits become partial (ds in-order completion).
// ---------------------------------------------------------------------------
__global__ __launch_bounds__(256)
void rel_kernel(const float* __restrict__ relF, const float* __restrict__ relBas,
                const unsigned char* __restrict__ maskb,
                const _Float16* __restrict__ wfrags_g,   // + biases appended (contiguous)
                const float* __restrict__ invNa, const float* __restrict__ pwpart,
                float* __restrict__ out) {
  extern __shared__ char smem[];
  _Float16* wlds = (_Float16*)smem;               // 68*1024 halves
  float* blds = (float*)(smem + 68*1024*2);       // 272 floats
  float* red  = blds + 272;                       // 8 waves * 12 floats

  const int tid  = threadIdx.x;
  const int wv   = tid >> 5;
  const int lane = tid & 31;
  const int bi   = blockIdx.x;
  const int b    = bi / NN;
  const int l15  = lane & 15;
  const int bofs = (lane >> 4) << 3;              // 0 (lanes 0-15) or 8 (lanes 16-31)

  // stage weights + biases (contiguous 140352B = 8772 float4) global -> LDS
  {
    const v4f* src = (const v4f*)wfrags_g;
    v4f* dst = (v4f*)smem;
    for (int idx = tid; idx < 8772; idx += 256) dst[idx] = src[idx];
  }
  __syncthreads();

  float rs[12];
  #pragma unroll
  for (int e = 0; e < 12; ++e) rs[e] = 0.0f;

  const size_t rowBase = (size_t)bi * NN;

  for (int tile = wv; tile < 32; tile += 8) {
    const int j0 = tile * 16;

    // ---- B frags of X^T : lane = batch row j0+l15, halves = features per A-layout
    H16 bx[4];
    const float* xp0 = relF + (rowBase + (size_t)(j0 + l15)) * FF + bofs;
    #pragma unroll
    for (int s = 0; s < 4; ++s) {
      const float* xp = xp0 + s*32;
      const v4f x0 = *(const v4f*)(xp);
      const v4f x1 = *(const v4f*)(xp + 4);
      const v4f x2 = *(const v4f*)(xp + 16);
      const v4f x3 = *(const v4f*)(xp + 20);
      bx[s].h2[0] = __builtin_amdgcn_cvt_pkrtz(x0[0], x0[1]);
      bx[s].h2[1] = __builtin_amdgcn_cvt_pkrtz(x0[2], x0[3]);
      bx[s].h2[2] = __builtin_amdgcn_cvt_pkrtz(x1[0], x1[1]);
      bx[s].h2[3] = __builtin_amdgcn_cvt_pkrtz(x1[2], x1[3]);
      bx[s].h2[4] = __builtin_amdgcn_cvt_pkrtz(x2[0], x2[1]);
      bx[s].h2[5] = __builtin_amdgcn_cvt_pkrtz(x2[2], x2[3]);
      bx[s].h2[6] = __builtin_amdgcn_cvt_pkrtz(x3[0], x3[1]);
      bx[s].h2[7] = __builtin_amdgcn_cvt_pkrtz(x3[2], x3[3]);
    }
    __builtin_amdgcn_sched_barrier(0);

    // ---- layer 1: H1^T[128 x 16] = W1^T X^T  (8 M-tiles x 4 K-steps)
    v8f acc[8] = {};
    #pragma unroll
    for (int s = 0; s < 4; ++s) {
      v16h a[8];
      #pragma unroll
      for (int t = 0; t < 8; ++t) a[t] = load_frag(wlds, t*4 + s, lane);
      __builtin_amdgcn_sched_barrier(0);   // all 16 ds loads issued before WMMAs
      #pragma unroll
      for (int t = 0; t < 8; ++t)
        acc[t] = __builtin_amdgcn_wmma_f32_16x16x32_f16(false, a[t], false, bx[s].v,
                                                        (short)0, acc[t], false, false);
      __builtin_amdgcn_sched_barrier(0);   // bound live A-frags to one K-step
    }

    // ---- epilogue 1: bias + ReLU + cvt -> layer-2 B frags (in registers, no transpose)
    H16 bh[4];
    #pragma unroll
    for (int s = 0; s < 4; ++s) {
      #pragma unroll
      for (int r = 0; r < 8; r += 2) {
        const float v0 = fmaxf(acc[2*s  ][r  ] + blds[(2*s  )*16 + r   + bofs], 0.0f);
        const float v1 = fmaxf(acc[2*s  ][r+1] + blds[(2*s  )*16 + r+1 + bofs], 0.0f);
        const float w0 = fmaxf(acc[2*s+1][r  ] + blds[(2*s+1)*16 + r   + bofs], 0.0f);
        const float w1 = fmaxf(acc[2*s+1][r+1] + blds[(2*s+1)*16 + r+1 + bofs], 0.0f);
        bh[s].h2[(r>>1)    ] = __builtin_amdgcn_cvt_pkrtz(v0, v1);
        bh[s].h2[(r>>1) + 4] = __builtin_amdgcn_cvt_pkrtz(w0, w1);
      }
    }
    __builtin_amdgcn_sched_barrier(0);

    // ---- layer 2
    #pragma unroll
    for (int t = 0; t < 8; ++t) acc[t] = (v8f){};
    #pragma unroll
    for (int s = 0; s < 4; ++s) {
      v16h a[8];
      #pragma unroll
      for (int t = 0; t < 8; ++t) a[t] = load_frag(wlds, 32 + t*4 + s, lane);
      __builtin_amdgcn_sched_barrier(0);
      #pragma unroll
      for (int t = 0; t < 8; ++t)
        acc[t] = __builtin_amdgcn_wmma_f32_16x16x32_f16(false, a[t], false, bh[s].v,
                                                        (short)0, acc[t], false, false);
      __builtin_amdgcn_sched_barrier(0);
    }

    // ---- epilogue 2 -> layer-3 B frags
    H16 b3f[4];
    #pragma unroll
    for (int s = 0; s < 4; ++s) {
      #pragma unroll
      for (int r = 0; r < 8; r += 2) {
        const float v0 = fmaxf(acc[2*s  ][r  ] + blds[128 + (2*s  )*16 + r   + bofs], 0.0f);
        const float v1 = fmaxf(acc[2*s  ][r+1] + blds[128 + (2*s  )*16 + r+1 + bofs], 0.0f);
        const float w0 = fmaxf(acc[2*s+1][r  ] + blds[128 + (2*s+1)*16 + r   + bofs], 0.0f);
        const float w1 = fmaxf(acc[2*s+1][r+1] + blds[128 + (2*s+1)*16 + r+1 + bofs], 0.0f);
        b3f[s].h2[(r>>1)    ] = __builtin_amdgcn_cvt_pkrtz(v0, v1);
        b3f[s].h2[(r>>1) + 4] = __builtin_amdgcn_cvt_pkrtz(w0, w1);
      }
    }
    __builtin_amdgcn_sched_barrier(0);

    // ---- layer 3: scale^T [16(pad from 4) x 16]
    v8f acc3 = {};
    {
      v16h a[4];
      #pragma unroll
      for (int s = 0; s < 4; ++s) a[s] = load_frag(wlds, 64 + s, lane);
      __builtin_amdgcn_sched_barrier(0);
      #pragma unroll
      for (int s = 0; s < 4; ++s)
        acc3 = __builtin_amdgcn_wmma_f32_16x16x32_f16(false, a[s], false, b3f[s].v,
                                                      (short)0, acc3, false, false);
    }
    __builtin_amdgcn_sched_barrier(0);

    // ---- masked basis contraction: lane n (<16) owns row j = j0+n, scales in acc3[0..3]
    if (lane < 16) {
      const int j = j0 + lane;
      const float mfac = maskb[(size_t)b*NN + j] ? 0.0f : 1.0f;
      const float* bp = relBas + (rowBase + (size_t)j) * (NB*3);
      const v4f q0 = *(const v4f*)(bp);
      const v4f q1 = *(const v4f*)(bp + 4);
      const v4f q2 = *(const v4f*)(bp + 8);
      const float sc0 = (acc3[0] + blds[256+0]) * mfac;
      const float sc1 = (acc3[1] + blds[256+1]) * mfac;
      const float sc2 = (acc3[2] + blds[256+2]) * mfac;
      const float sc3 = (acc3[3] + blds[256+3]) * mfac;
      rs[0] += sc0*q0[0]; rs[1]  += sc0*q0[1]; rs[2]  += sc0*q0[2];
      rs[3] += sc1*q0[3]; rs[4]  += sc1*q1[0]; rs[5]  += sc1*q1[1];
      rs[6] += sc2*q1[2]; rs[7]  += sc2*q1[3]; rs[8]  += sc2*q2[0];
      rs[9] += sc3*q2[1]; rs[10] += sc3*q2[2]; rs[11] += sc3*q2[3];
    }
    __builtin_amdgcn_sched_barrier(0);
  }

  // ---- wave reduce (lanes >=16 hold zeros) then workgroup reduce
  #pragma unroll
  for (int e = 0; e < 12; ++e) {
    float v = rs[e];
    for (int off = 16; off >= 1; off >>= 1) v += __shfl_xor(v, off, 32);
    rs[e] = v;
  }
  if (lane == 0) {
    #pragma unroll
    for (int e = 0; e < 12; ++e) red[wv*12 + e] = rs[e];
  }
  __syncthreads();
  if (tid < 3) {
    float s = 0.0f;
    for (int w = 0; w < 8; ++w)
      for (int k = 0; k < NB; ++k)
        s += red[w*12 + k*3 + tid];
    out[bi*3 + tid] = pwpart[bi*3 + tid] + invNa[2 + b] * s;
  }
}

// ---------------------------------------------------------------------------
extern "C" void kernel_launch(void* const* d_in, const int* in_sizes, int n_in,
                              void* d_out, int out_size, void* d_ws, size_t ws_size,
                              hipStream_t stream) {
  (void)in_sizes; (void)n_in; (void)out_size; (void)ws_size;
  const float* pwF    = (const float*)d_in[0];
  const float* relF   = (const float*)d_in[1];
  const float* pwBas  = (const float*)d_in[2];
  const float* relBas = (const float*)d_in[3];
  const float* pwW1   = (const float*)d_in[4];
  const float* pwb1   = (const float*)d_in[5];
  const float* pwW2   = (const float*)d_in[6];
  const float* pwb2   = (const float*)d_in[7];
  const float* pwW3   = (const float*)d_in[8];
  const float* pwb3   = (const float*)d_in[9];
  const float* reW1   = (const float*)d_in[10];
  const float* reb1   = (const float*)d_in[11];
  const float* reW2   = (const float*)d_in[12];
  const float* reb2   = (const float*)d_in[13];
  const float* reW3   = (const float*)d_in[14];
  const float* reb3   = (const float*)d_in[15];
  const unsigned char* maskb = (const unsigned char*)d_in[16];

  char* ws = (char*)d_ws;
  _Float16* wfrags = (_Float16*)(ws + WS_WFRAGS);
  float* biases    = (float*)(ws + WS_BIAS);
  float* invNa     = (float*)(ws + WS_INVNA);
  float* pwpart    = (float*)(ws + WS_PWPART);

  prep_kernel<<<70, 256, 0, stream>>>(reW1, reW2, reW3, reb1, reb2, reb3,
                                      maskb, wfrags, biases, invNa);
  pw_kernel<<<BB*NN, 128, 0, stream>>>(pwF, pwBas, pwW1, pwb1, pwW2, pwb2,
                                       pwW3, pwb3, invNa, pwpart);
  const size_t shmem = (size_t)68*1024*2 + 272*4 + 8*12*4;  // 140736 B dynamic LDS
  rel_kernel<<<BB*NN, 256, shmem, stream>>>(relF, relBas, maskb, wfrags,
                                            invNa, pwpart, (float*)d_out);
}
